// MultiHeadedAttention_21706764714611
// MI455X (gfx1250) — compile-verified, matrix operand order checked by reading, not run on hardware
//
#include <hip/hip_runtime.h>
#include <hip/hip_bf16.h>
#include <stdint.h>

// ---------------- config ----------------
#define B_   2
#define T_   2048
#define C_   2048
#define H_   16
#define DH_  64
#define BT_  (B_ * T_)          // 4096 rows
#define CK_  (C_ / 4)           // 512 (K/V projection width)

static constexpr float LAMBDA_INIT_ = 0.35550906759096928f; // 0.8 - 0.6*exp(-0.3)
static constexpr float SCALE_       = 0.125f;               // 1/sqrt(64)
static constexpr float EPS_         = 1e-6f;

// ---------------- types ----------------
typedef __attribute__((ext_vector_type(8)))  float  floatx8;
typedef __attribute__((ext_vector_type(8)))  __bf16 bf16x8;
typedef __attribute__((ext_vector_type(16))) __bf16 bf16x16;

union BF16x16 { bf16x16 v; bf16x8 h[2]; };

__device__ __forceinline__ floatx8 wmma_bf16(BF16x16 a, BF16x16 b, floatx8 c) {
  // 8 args: (neg_a, A, neg_b, B, c_mod, C, reuse_a, reuse_b)
  return __builtin_amdgcn_wmma_f32_16x16x32_bf16(false, a.v, false, b.v,
                                                 (short)0, c, false, false);
}

// async copy 16B global -> LDS (ASYNCcnt-tracked, no VGPR data)
__device__ __forceinline__ void async_copy_b128(uint32_t lds_off, const void* gptr) {
  asm volatile("global_load_async_to_lds_b128 %0, %1, off"
               :: "v"(lds_off), "v"(gptr) : "memory");
}
__device__ __forceinline__ void wait_asynccnt0() {
  asm volatile("s_wait_asynccnt 0x0" ::: "memory");
}
__device__ __forceinline__ void wait_dscnt0() {
  asm volatile("s_wait_dscnt 0x0" ::: "memory");
}

// ---------------- f32 -> bf16 convert ----------------
__global__ __launch_bounds__(256)
void cvt_f32_bf16(const float* __restrict__ in, __bf16* __restrict__ out, size_t n) {
  size_t i = (size_t)blockIdx.x * blockDim.x + threadIdx.x;
  size_t stride = (size_t)gridDim.x * blockDim.x;
  for (; i < n; i += stride) out[i] = (__bf16)in[i];
}

// ---------------- bf16 WMMA GEMM: C[M,N] = A[M,K] * W[N,K]^T (+bias) ----------------
// block = 128 threads (4 waves in 2x2), each wave computes 64x64 (4x4 WMMA tiles).
// A-matrix 16-bit layout: lane l (row = l&15, half h = l>>4) packs K = {8h.., 16+8h..}.
__global__ __launch_bounds__(128)
void gemm_bf16_kernel(const __bf16* __restrict__ A, const __bf16* __restrict__ W,
                      const float* __restrict__ bias, float* __restrict__ Cc,
                      int M, int N, int K) {
  const int lane = threadIdx.x & 31;
  const int wave = threadIdx.x >> 5;
  const int r  = lane & 15;
  const int hh = lane >> 4;
  const int m0 = blockIdx.y * 128 + (wave >> 1) * 64;
  const int n0 = blockIdx.x * 128 + (wave & 1) * 64;

  const __bf16* arow[4];
  const __bf16* brow[4];
#pragma unroll
  for (int i = 0; i < 4; ++i) {
    arow[i] = A + (size_t)(m0 + i * 16 + r) * K;
    brow[i] = W + (size_t)(n0 + i * 16 + r) * K;
  }

  floatx8 acc[4][4];
#pragma unroll
  for (int mi = 0; mi < 4; ++mi)
#pragma unroll
    for (int ni = 0; ni < 4; ++ni)
      acc[mi][ni] = (floatx8){0.f,0.f,0.f,0.f,0.f,0.f,0.f,0.f};

  for (int k0 = 0; k0 < K; k0 += 32) {
    __builtin_prefetch(arow[0] + k0 + 512, 0, 1);   // global_prefetch_b8
    __builtin_prefetch(brow[0] + k0 + 512, 0, 1);
    BF16x16 af[4], bf[4];
#pragma unroll
    for (int i = 0; i < 4; ++i) {
      af[i].h[0] = *(const bf16x8*)(arow[i] + k0 +      8 * hh);
      af[i].h[1] = *(const bf16x8*)(arow[i] + k0 + 16 + 8 * hh);
      bf[i].h[0] = *(const bf16x8*)(brow[i] + k0 +      8 * hh);
      bf[i].h[1] = *(const bf16x8*)(brow[i] + k0 + 16 + 8 * hh);
    }
#pragma unroll
    for (int mi = 0; mi < 4; ++mi)
#pragma unroll
      for (int ni = 0; ni < 4; ++ni)
        acc[mi][ni] = wmma_bf16(af[mi], bf[ni], acc[mi][ni]);
  }

  // C/D layout: vgpr i, lane l -> M = i + 8*hh, N = l&15
#pragma unroll
  for (int ni = 0; ni < 4; ++ni) {
    const int col = n0 + ni * 16 + r;
    const float bb = bias ? bias[col] : 0.0f;
#pragma unroll
    for (int mi = 0; mi < 4; ++mi) {
      float* crow = Cc + (size_t)(m0 + mi * 16 + 8 * hh) * N + col;
#pragma unroll
      for (int i = 0; i < 8; ++i) crow[(size_t)i * N] = acc[mi][ni][i] + bb;
    }
  }
}

// ---------------- RoPE + convert to bf16 (q: HS=32 subheads, k: HS=8) ----------------
__global__ __launch_bounds__(256)
void rope_cvt_kernel(const float* __restrict__ x, __bf16* __restrict__ out, int HS) {
  size_t idx = (size_t)blockIdx.x * blockDim.x + threadIdx.x;
  size_t total = (size_t)BT_ * HS * 32;   // pairs
  if (idx >= total) return;
  int j = (int)(idx & 31);
  size_t tmp = idx >> 5;
  int s = (int)(tmp % HS);
  size_t row = tmp / HS;                  // b*T + t
  int t = (int)(row % T_);
  float freq = (float)t * __powf(10000.0f, -(float)j * (1.0f / 32.0f));
  float cs = cosf(freq), sn = sinf(freq);
  size_t base = (row * HS + s) * 64 + 2 * j;
  float x0 = x[base], x1 = x[base + 1];
  out[base]     = (__bf16)(x0 * cs - x1 * sn);
  out[base + 1] = (__bf16)(x0 * sn + x1 * cs);
}

// ---------------- V transpose + convert: [B,T,4,128] f32 -> [B,4,128,T] bf16 ----------------
__global__ __launch_bounds__(256)
void vt_cvt_kernel(const float* __restrict__ v, __bf16* __restrict__ vt) {
  size_t idx = (size_t)blockIdx.x * blockDim.x + threadIdx.x;
  size_t total = (size_t)BT_ * 512;
  if (idx >= total) return;
  int d  = (int)(idx & 127);
  int hk = (int)((idx >> 7) & 3);
  size_t row = idx >> 9;                 // b*T + t
  int t = (int)(row % T_);
  int b = (int)(row / T_);
  vt[(((size_t)b * 4 + hk) * 128 + d) * T_ + t] = (__bf16)v[idx];
}

// ---------------- lambda per head ----------------
__global__ void lam_kernel(const float* __restrict__ lq1, const float* __restrict__ lk1,
                           const float* __restrict__ lq2, const float* __restrict__ lk2,
                           float* __restrict__ lam) {
  int h = threadIdx.x;
  if (h >= H_) return;
  float s1 = 0.f, s2 = 0.f;
  for (int d = 0; d < DH_; ++d) {
    s1 += lq1[h * DH_ + d] * lk1[h * DH_ + d];
    s2 += lq2[h * DH_ + d] * lk2[h * DH_ + d];
  }
  lam[h] = __expf(s1) - __expf(s2);   // LAMBDA_INIT folded in at combine
}

// ---------------- flash attention per sub-head (online softmax, WMMA S and PV) ----------------
// Qb: [B,T,32,64] bf16 ; Kb: [B,T,8,64] bf16 ; VTb: [B,4,128,T] bf16
// Osub: [B,32,T,128] f32. Block = 4 waves, 64 query rows, shared async-staged K/V tiles.
__global__ __launch_bounds__(128)
void flash_diff_kernel(const __bf16* __restrict__ Qb, const __bf16* __restrict__ Kb,
                       const __bf16* __restrict__ VTb, float* __restrict__ Osub) {
  __shared__ __bf16 kt[32 * 64];     // 4KB : [key 0..31][dh 0..63]
  __shared__ __bf16 vt[128 * 32];    // 8KB : [dv 0..127][key 0..31]
  __shared__ __bf16 plds[4 * 512];   // per-wave 16x32 bf16 P bounce buffers

  const int tid  = threadIdx.x;
  const int lane = tid & 31;
  const int wave = tid >> 5;
  const int r  = lane & 15;
  const int hh = lane >> 4;
  const int hq = blockIdx.y;               // 0..31 sub-head
  const int b  = blockIdx.z;
  const int qbase = (blockIdx.x * 4 + wave) * 16;
  const int qhi = qbase + 15;
  const int block_qhi = blockIdx.x * 64 + 63;
  const int ksub = hq >> 2;                // GQA: k sub-head
  const int vkv  = hq >> 3;                // kv head for V

  const uint32_t kt_base = (uint32_t)(uintptr_t)(&kt[0]);
  const uint32_t vt_base = (uint32_t)(uintptr_t)(&vt[0]);

  // Q rows for this wave's 16 queries, A-matrix layout for dh[0..31] and dh[32..63]
  const __bf16* qrow = Qb + (((size_t)(b * T_ + qbase + r)) * 32 + hq) * 64;
  BF16x16 qa0, qa1;
  qa0.h[0] = *(const bf16x8*)(qrow +      8 * hh);
  qa0.h[1] = *(const bf16x8*)(qrow + 16 + 8 * hh);
  qa1.h[0] = *(const bf16x8*)(qrow + 32 + 8 * hh);
  qa1.h[1] = *(const bf16x8*)(qrow + 48 + 8 * hh);

  floatx8 o[8];
#pragma unroll
  for (int n = 0; n < 8; ++n) o[n] = (floatx8){0.f,0.f,0.f,0.f,0.f,0.f,0.f,0.f};
  float mrow[8], lrow[8];
#pragma unroll
  for (int i = 0; i < 8; ++i) { mrow[i] = -3.0e38f; lrow[i] = 0.f; }

  __bf16* myp = plds + wave * 512;

  // K/V staging addresses (per-thread, shifted by s0 each iteration)
  const int krow = tid >> 2;               // 0..31
  const int kpp  = (tid & 3) * 2;          // part pairs
  const __bf16* kg_base = Kb + (((size_t)(b * T_ + krow)) * 8 + ksub) * 64;
  const __bf16* vg_base = VTb + (((size_t)(b * 4 + vkv)) * 128 + tid) * T_;

  for (int s0 = 0; s0 <= block_qhi; s0 += 32) {
    // ---- async-stage shared K tile (32x64) and V tile (128x32) into LDS ----
#pragma unroll
    for (int j = 0; j < 2; ++j) {
      const int part = kpp + j;            // 0..7, 16B each
      async_copy_b128(kt_base + (uint32_t)(krow * 128 + part * 16),
                      kg_base + (size_t)s0 * (8 * 64) + part * 8);
    }
#pragma unroll
    for (int j = 0; j < 4; ++j) {          // dv = tid, 4x16B per row
      async_copy_b128(vt_base + (uint32_t)(tid * 64 + j * 16),
                      vg_base + s0 + j * 8);
    }
    wait_asynccnt0();
    __syncthreads();

    if (s0 <= qhi) {
      // ---- S = Q K^T for two 16-key tiles, K B-fragments from LDS ----
      const int key0 = s0 + r;
      const int key1 = s0 + 16 + r;
      floatx8 sa = {0.f,0.f,0.f,0.f,0.f,0.f,0.f,0.f};
      floatx8 sb = {0.f,0.f,0.f,0.f,0.f,0.f,0.f,0.f};
      BF16x16 kf;
      const __bf16* k0l = kt + r * 64;
      const __bf16* k1l = kt + (16 + r) * 64;
      kf.h[0] = *(const bf16x8*)(k0l +      8 * hh);
      kf.h[1] = *(const bf16x8*)(k0l + 16 + 8 * hh);
      sa = wmma_bf16(qa0, kf, sa);
      kf.h[0] = *(const bf16x8*)(k0l + 32 + 8 * hh);
      kf.h[1] = *(const bf16x8*)(k0l + 48 + 8 * hh);
      sa = wmma_bf16(qa1, kf, sa);
      kf.h[0] = *(const bf16x8*)(k1l +      8 * hh);
      kf.h[1] = *(const bf16x8*)(k1l + 16 + 8 * hh);
      sb = wmma_bf16(qa0, kf, sb);
      kf.h[0] = *(const bf16x8*)(k1l + 32 + 8 * hh);
      kf.h[1] = *(const bf16x8*)(k1l + 48 + 8 * hh);
      sb = wmma_bf16(qa1, kf, sb);

      // ---- online softmax update (row stats across the 16-lane half) ----
      float fac[8];
#pragma unroll
      for (int i = 0; i < 8; ++i) {
        const int qr = qbase + i + 8 * hh;
        float x0 = sa[i] * SCALE_; if (key0 > qr) x0 = -3.0e38f;
        float x1 = sb[i] * SCALE_; if (key1 > qr) x1 = -3.0e38f;
        float mx = fmaxf(x0, x1);
        mx = fmaxf(mx, __shfl_xor(mx, 1, 32));
        mx = fmaxf(mx, __shfl_xor(mx, 2, 32));
        mx = fmaxf(mx, __shfl_xor(mx, 4, 32));
        mx = fmaxf(mx, __shfl_xor(mx, 8, 32));
        const float mnew = fmaxf(mrow[i], mx);
        const float f = __expf(mrow[i] - mnew);
        const float e0 = __expf(x0 - mnew);
        const float e1 = __expf(x1 - mnew);
        float rs = e0 + e1;
        rs += __shfl_xor(rs, 1, 32);
        rs += __shfl_xor(rs, 2, 32);
        rs += __shfl_xor(rs, 4, 32);
        rs += __shfl_xor(rs, 8, 32);
        lrow[i] = lrow[i] * f + rs;
        mrow[i] = mnew;
        fac[i] = f;
        // stash P row-major 16x32 bf16 (C-layout -> LDS)
        myp[(i + 8 * hh) * 32 + r]      = (__bf16)e0;
        myp[(i + 8 * hh) * 32 + 16 + r] = (__bf16)e1;
      }
#pragma unroll
      for (int n = 0; n < 8; ++n)
#pragma unroll
        for (int i = 0; i < 8; ++i) o[n][i] *= fac[i];

      wait_dscnt0();

      // reload P in A-matrix layout (per-wave private region)
      BF16x16 pa;
      pa.h[0] = *(const bf16x8*)(myp + r * 32 +      8 * hh);
      pa.h[1] = *(const bf16x8*)(myp + r * 32 + 16 + 8 * hh);

      // ---- O += P * V, 8 N-tiles of dv, V B-fragments from LDS ----
#pragma unroll
      for (int n = 0; n < 8; ++n) {
        const __bf16* vrow = vt + (n * 16 + r) * 32;
        BF16x16 vf;
        vf.h[0] = *(const bf16x8*)(vrow +      8 * hh);
        vf.h[1] = *(const bf16x8*)(vrow + 16 + 8 * hh);
        o[n] = wmma_bf16(pa, vf, o[n]);
      }
    }

    wait_dscnt0();
    __syncthreads();   // all LDS reads done before next iteration's async stores
  }

  // ---- epilogue: normalize and store [b,hq,q,dv] ----
  float* obase = Osub + (((size_t)(b * 32 + hq)) * T_ + qbase + 8 * hh) * 128 + r;
#pragma unroll
  for (int n = 0; n < 8; ++n)
#pragma unroll
    for (int i = 0; i < 8; ++i)
      obase[(size_t)i * 128 + n * 16] = o[n][i] / lrow[i];
}

// ---------------- combine sub-heads + RMSNorm -> yb bf16 [B,T,C] ----------------
__global__ __launch_bounds__(128)
void combine_kernel(const float* __restrict__ osub, const float* __restrict__ lam,
                    const float* __restrict__ dnw, __bf16* __restrict__ yb) {
  const int lane = threadIdx.x & 31;
  const int wave = threadIdx.x >> 5;
  size_t row = (size_t)blockIdx.x * 4 + wave;    // over B*H*T
  const int t = (int)(row % T_);
  const int h = (int)((row / T_) % H_);
  const int b = (int)(row / ((size_t)T_ * H_));
  const float* o1 = osub + (((size_t)(b * 32 + 2 * h))     * T_ + t) * 128;
  const float* o2 = osub + (((size_t)(b * 32 + 2 * h + 1)) * T_ + t) * 128;
  const float lm = lam[h] + LAMBDA_INIT_;
  float y[4]; float ss = 0.f;
#pragma unroll
  for (int j = 0; j < 4; ++j) {
    const int d = lane + 32 * j;
    y[j] = o1[d] - lm * o2[d];
    ss += y[j] * y[j];
  }
  ss += __shfl_xor(ss, 1, 32);
  ss += __shfl_xor(ss, 2, 32);
  ss += __shfl_xor(ss, 4, 32);
  ss += __shfl_xor(ss, 8, 32);
  ss += __shfl_xor(ss, 16, 32);
  const float rn = rsqrtf(ss * (1.0f / 128.0f) + EPS_) * (1.0f - LAMBDA_INIT_);
  __bf16* yrow = yb + ((size_t)(b * T_ + t)) * C_ + h * 128;
#pragma unroll
  for (int j = 0; j < 4; ++j) {
    const int d = lane + 32 * j;
    yrow[d] = (__bf16)(y[j] * rn * dnw[d]);
  }
}

// ---------------- host launch ----------------
extern "C" void kernel_launch(void* const* d_in, const int* in_sizes, int n_in,
                              void* d_out, int out_size, void* d_ws, size_t ws_size,
                              hipStream_t stream) {
  const float* xq  = (const float*)d_in[0];
  const float* xk  = (const float*)d_in[1];
  const float* xv  = (const float*)d_in[2];
  // d_in[3..6]: causal_mask / start_pos / inference / flash_attn (baked)
  const float* Wq  = (const float*)d_in[7];
  const float* bq  = (const float*)d_in[8];
  const float* Wk  = (const float*)d_in[9];
  const float* bk  = (const float*)d_in[10];
  const float* Wv  = (const float*)d_in[11];
  const float* bv  = (const float*)d_in[12];
  const float* Wo  = (const float*)d_in[13];
  const float* lq1 = (const float*)d_in[14];
  const float* lk1 = (const float*)d_in[15];
  const float* lq2 = (const float*)d_in[16];
  const float* lk2 = (const float*)d_in[17];
  const float* dnw = (const float*)d_in[18];

  char* p = (char*)d_ws;
  auto alloc = [&](size_t bytes) -> void* {
    void* r = (void*)p; p += (bytes + 255) & ~(size_t)255; return r;
  };
  __bf16* xq_b = (__bf16*)alloc((size_t)BT_ * C_  * 2);
  __bf16* xk_b = (__bf16*)alloc((size_t)BT_ * C_  * 2);
  __bf16* xv_b = (__bf16*)alloc((size_t)BT_ * C_  * 2);
  __bf16* Wq_b = (__bf16*)alloc((size_t)C_  * C_  * 2);
  __bf16* Wk_b = (__bf16*)alloc((size_t)CK_ * C_  * 2);
  __bf16* Wv_b = (__bf16*)alloc((size_t)CK_ * C_  * 2);
  __bf16* Wo_b = (__bf16*)alloc((size_t)C_  * C_  * 2);
  float*  qf   = (float*) alloc((size_t)BT_ * C_  * 4);
  float*  kf   = (float*) alloc((size_t)BT_ * CK_ * 4);
  float*  vf   = (float*) alloc((size_t)BT_ * CK_ * 4);
  __bf16* qb   = (__bf16*)alloc((size_t)BT_ * C_  * 2);
  __bf16* kb   = (__bf16*)alloc((size_t)BT_ * CK_ * 2);
  __bf16* vtb  = (__bf16*)alloc(((size_t)B_ * 4 * 128 * T_ + 64) * 2);
  float*  osub = (float*) alloc((size_t)B_ * 32 * T_ * 128 * 4);
  __bf16* yb   = (__bf16*)alloc((size_t)BT_ * C_  * 2);
  float*  lamw = (float*) alloc(64 * 4);

  auto cvt = [&](const float* in, __bf16* out, size_t n) {
    int blocks = (int)((n + 2047) / 2048);
    if (blocks > 8192) blocks = 8192;
    cvt_f32_bf16<<<blocks, 256, 0, stream>>>(in, out, n);
  };
  cvt(xq, xq_b, (size_t)BT_ * C_);
  cvt(xk, xk_b, (size_t)BT_ * C_);
  cvt(xv, xv_b, (size_t)BT_ * C_);
  cvt(Wq, Wq_b, (size_t)C_  * C_);
  cvt(Wk, Wk_b, (size_t)CK_ * C_);
  cvt(Wv, Wv_b, (size_t)CK_ * C_);
  cvt(Wo, Wo_b, (size_t)C_  * C_);

  // QKV projections (WMMA, 128x128 per block / 64x64 per wave)
  gemm_bf16_kernel<<<dim3(C_  / 128, BT_ / 128), 128, 0, stream>>>(xq_b, Wq_b, bq, qf, BT_, C_,  C_);
  gemm_bf16_kernel<<<dim3(CK_ / 128, BT_ / 128), 128, 0, stream>>>(xk_b, Wk_b, bk, kf, BT_, CK_, C_);
  gemm_bf16_kernel<<<dim3(CK_ / 128, BT_ / 128), 128, 0, stream>>>(xv_b, Wv_b, bv, vf, BT_, CK_, C_);

  // RoPE + bf16 ; V transpose + bf16
  {
    size_t nq = (size_t)BT_ * 32 * 32;   // pairs
    rope_cvt_kernel<<<(unsigned)((nq + 255) / 256), 256, 0, stream>>>(qf, qb, 32);
    size_t nk = (size_t)BT_ * 8 * 32;
    rope_cvt_kernel<<<(unsigned)((nk + 255) / 256), 256, 0, stream>>>(kf, kb, 8);
    size_t nv = (size_t)BT_ * 512;
    vt_cvt_kernel<<<(unsigned)((nv + 255) / 256), 256, 0, stream>>>(vf, vtb);
  }

  lam_kernel<<<1, 16, 0, stream>>>(lq1, lk1, lq2, lk2, lamw);

  // Flash differential attention: 32 sub-heads, 64 query rows per block
  flash_diff_kernel<<<dim3(T_ / 64, 32, B_), 128, 0, stream>>>(qb, kb, vtb, osub);

  // Combine halves + RMSNorm -> bf16 [B,T,C]
  combine_kernel<<<(unsigned)((size_t)B_ * H_ * T_ / 4), 128, 0, stream>>>(osub, lamw, dnw, yb);

  // Output projection (WMMA) -> d_out (f32)
  gemm_bf16_kernel<<<dim3(C_ / 128, BT_ / 128), 128, 0, stream>>>(yb, Wo_b, nullptr,
                                                                  (float*)d_out, BT_, C_, C_);
}
